// ContMix_11605001634455
// MI455X (gfx1250) — matrix operand order repaired and checked by reading, not compiled, not run on hardware
//
#include <hip/hip_runtime.h>
#include <math.h>

typedef __attribute__((ext_vector_type(16))) _Float16 v16h;
typedef __attribute__((ext_vector_type(8)))  float    v8f;

// ---------------- problem constants ----------------
constexpr int B_    = 16;
constexpr int DIM_  = 384;
constexpr int H_    = 56;
constexpr int W_    = 56;
constexpr int HW_   = H_ * W_;
constexpr int CHALF = 192;   // DIM/2
constexpr int HEADS = 4;
constexpr int CH_   = 48;    // CHALF/HEADS
constexpr int KBIG  = 7;
constexpr int KSM   = 5;
constexpr int NTAP  = KBIG*KBIG + KSM*KSM;  // 74
constexpr int NPAD  = 80;                   // padded attn columns
constexpr float EPSV = 1e-5f;

#define DEV __device__ __forceinline__

// =====================================================================
// Packed fragment layouts (wave32 16-bit operands, contiguous k = 16*hlf + elem)
//   A tile = 16(M) x 32(K) f16, B tile = 32(K) x 16(N) f16.
//   Storage: [batch][tileM or tileN][tileK][lane 0..31][elem 0..15]
//   A: lane = hlf*16 + m%16 ; k_local = (hlf<<4)|elem
//   B: lane = hlf*16 + n%16 ; k_local = (hlf<<4)|elem
// =====================================================================

__global__ void pack_a_kernel(const float* __restrict__ A,
                              long sAm, long sAk, long sAb,
                              _Float16* __restrict__ out,
                              int M, int K, int MT, int KT, long total)
{
    long idx = (long)blockIdx.x * blockDim.x + threadIdx.x;
    if (idx >= total) return;
    int elem = (int)(idx & 15); long t = idx >> 4;
    int lane = (int)(t & 31);  t >>= 5;
    int kt = (int)(t % KT);    t /= KT;
    int mt = (int)(t % MT);    int b = (int)(t / MT);
    int hlf = lane >> 4, l16 = lane & 15;
    int kl = (hlf << 4) | elem;
    int m = mt * 16 + l16;
    int k = kt * 32 + kl;
    float v = (k < K && m < M) ? A[(long)b * sAb + (long)m * sAm + (long)k * sAk] : 0.f;
    out[idx] = (_Float16)v;
}

__global__ void pack_b_kernel(const float* __restrict__ Bsrc,
                              long sBk, long sBn, long sBb,
                              _Float16* __restrict__ out,
                              int N, int K, int NT4, int KT, long total)
{
    long idx = (long)blockIdx.x * blockDim.x + threadIdx.x;
    if (idx >= total) return;
    int elem = (int)(idx & 15); long t = idx >> 4;
    int lane = (int)(t & 31);  t >>= 5;
    int kt = (int)(t % KT);    t /= KT;
    int nt = (int)(t % NT4);   int b = (int)(t / NT4);
    int hlf = lane >> 4, l16 = lane & 15;
    int kl = (hlf << 4) | elem;
    int n = nt * 16 + l16;
    int k = kt * 32 + kl;
    float v = (k < K && n < N) ? Bsrc[(long)b * sBb + (long)k * sBk + (long)n * sBn] : 0.f;
    out[idx] = (_Float16)v;
}

// =====================================================================
// Packed-operand WMMA GEMM. Wave computes 32(M) x 64(N) (8 accumulators,
// B fragments reused across 2 A fragments); block = 2 waves -> 64 x 64.
// Inner loop: 2 A loads + 4 B loads + 8 WMMAs, no guards, no divergence.
// Epilogue: optional per-row BN scale/bias, scalar multiplier, addend.
// =====================================================================
__global__ void wmma_gemm_packed(
    const _Float16* __restrict__ Ap, long aBatchTiles,  // MT*KT (0 => shared A)
    const _Float16* __restrict__ Bp, long bBatchTiles,  // NT4*KT
    int KT,
    float* __restrict__ C, long sCm, long sCn, long sCb,
    const float* __restrict__ rowScale, const float* __restrict__ rowBias,
    float bnR, float outMul,
    const float* __restrict__ addend, long sDb,
    int M, int N)
{
    const int lane = threadIdx.x & 31;
    const int wave = threadIdx.x >> 5;
    const int hlf  = lane >> 4;
    const int l16  = lane & 15;
    const int mt0  = (blockIdx.y * 2 + wave) * 2;   // first of two M-tiles
    const int nt0  = blockIdx.x * 4;
    const long bb  = blockIdx.z;

    const v16h* pA0 = (const v16h*)Ap + (bb * aBatchTiles + (long)(mt0 + 0) * KT) * 32 + lane;
    const v16h* pA1 = pA0 + (long)KT * 32;
    const v16h* pB0 = (const v16h*)Bp + (bb * bBatchTiles + (long)(nt0 + 0) * KT) * 32 + lane;
    const v16h* pB1 = pB0 + (long)KT * 32;
    const v16h* pB2 = pB1 + (long)KT * 32;
    const v16h* pB3 = pB2 + (long)KT * 32;

    const v8f vz = {0.f,0.f,0.f,0.f,0.f,0.f,0.f,0.f};
    v8f acc[2][4];
    #pragma unroll
    for (int u = 0; u < 2; ++u)
        #pragma unroll
        for (int t = 0; t < 4; ++t) acc[u][t] = vz;

    #pragma unroll 2
    for (int kt = 0; kt < KT; ++kt) {
        const long o = (long)kt * 32;
        __builtin_prefetch(pA0 + o + 32, 0, 1);      // next A fragment stream
        v16h a0 = pA0[o];
        v16h a1 = pA1[o];
        v16h b0 = pB0[o];
        v16h b1 = pB1[o];
        v16h b2 = pB2[o];
        v16h b3 = pB3[o];
        acc[0][0] = __builtin_amdgcn_wmma_f32_16x16x32_f16(false, a0, false, b0, (short)0, acc[0][0], false, false);
        acc[0][1] = __builtin_amdgcn_wmma_f32_16x16x32_f16(false, a0, false, b1, (short)0, acc[0][1], false, false);
        acc[0][2] = __builtin_amdgcn_wmma_f32_16x16x32_f16(false, a0, false, b2, (short)0, acc[0][2], false, false);
        acc[0][3] = __builtin_amdgcn_wmma_f32_16x16x32_f16(false, a0, false, b3, (short)0, acc[0][3], false, false);
        acc[1][0] = __builtin_amdgcn_wmma_f32_16x16x32_f16(false, a1, false, b0, (short)0, acc[1][0], false, false);
        acc[1][1] = __builtin_amdgcn_wmma_f32_16x16x32_f16(false, a1, false, b1, (short)0, acc[1][1], false, false);
        acc[1][2] = __builtin_amdgcn_wmma_f32_16x16x32_f16(false, a1, false, b2, (short)0, acc[1][2], false, false);
        acc[1][3] = __builtin_amdgcn_wmma_f32_16x16x32_f16(false, a1, false, b3, (short)0, acc[1][3], false, false);
    }

    // ---- epilogue + store (C/D layout: VGPR r -> M = r + 8*hlf) ----
    #pragma unroll
    for (int u = 0; u < 2; ++u) {
        #pragma unroll
        for (int r = 0; r < 8; ++r) {
            int m = (mt0 + u) * 16 + (hlf << 3) + r;
            float s = 1.f, o = 0.f;
            if (rowScale) { s = rowScale[m] * bnR; o = rowBias[m]; }
            #pragma unroll
            for (int t = 0; t < 4; ++t) {
                int n = (nt0 + t) * 16 + l16;
                if (n < N) {
                    float v = acc[u][t][r];
                    v = v * s + o;
                    v *= outMul;
                    long ci = bb * sCb + (long)m * sCm + (long)n * sCn;
                    if (addend) v += addend[bb * sDb + (long)m * sCm + (long)n * sCn];
                    C[ci] = v;
                }
            }
        }
    }
}

// =====================================================================
// lepe: fused 5-branch dilated depthwise re-param block + BNs
// =====================================================================
DEV float dwconv_at(const float* __restrict__ xc, const float* __restrict__ w,
                    int y, int x, int k, int dil)
{
    int pad = (dil * (k - 1) + 1) / 2;
    float s = 0.f;
    for (int i = 0; i < k; ++i) {
        int yy = y + i * dil - pad;
        if (yy < 0 || yy >= H_) continue;
        for (int j = 0; j < k; ++j) {
            int xx = x + j * dil - pad;
            if (xx < 0 || xx >= W_) continue;
            s += w[i * k + j] * xc[yy * W_ + xx];
        }
    }
    return s;
}

__global__ void lepe_kernel(
    const float* __restrict__ x,
    const float* __restrict__ lkw,
    const float* __restrict__ g0, const float* __restrict__ b0,
    const float* __restrict__ dw0, const float* __restrict__ g1, const float* __restrict__ b1,
    const float* __restrict__ dw1, const float* __restrict__ g2, const float* __restrict__ b2,
    const float* __restrict__ dw2, const float* __restrict__ g3, const float* __restrict__ b3,
    const float* __restrict__ dw3, const float* __restrict__ g4, const float* __restrict__ b4,
    const float* __restrict__ go,  const float* __restrict__ bo,
    float* __restrict__ out)
{
    long idx = (long)blockIdx.x * blockDim.x + threadIdx.x;
    if (idx >= (long)B_ * DIM_ * HW_) return;
    int xp = (int)(idx % W_); long t = idx / W_;
    int yp = (int)(t % H_);  t /= H_;
    int ch = (int)(t % DIM_); int b = (int)(t / DIM_);

    const float* xc = x + ((long)(b * DIM_ + ch)) * HW_;
    const float invs = rsqrtf(1.f + EPSV);
    const float sout = go[ch] * invs;

    float c0 = dwconv_at(xc, lkw + ch * 49, yp, xp, 7, 1);
    float c1 = dwconv_at(xc, dw0 + ch * 25, yp, xp, 5, 1);
    float c2 = dwconv_at(xc, dw1 + ch * 9,  yp, xp, 3, 1);
    float c3 = dwconv_at(xc, dw2 + ch * 9,  yp, xp, 3, 2);
    float c4 = dwconv_at(xc, dw3 + ch * 9,  yp, xp, 3, 3);

    float acc = c0 * (g0[ch] * invs) + c1 * (g1[ch] * invs) + c2 * (g2[ch] * invs)
              + c3 * (g3[ch] * invs) + c4 * (g4[ch] * invs);
    float bsum = b0[ch] + b1[ch] + b2[ch] + b3[ch] + b4[ch];
    out[idx] = sout * (acc + bsum) + bo[ch];
}

// =====================================================================
// AdaptiveAvgPool2d(7) : (B,192,56,56) -> (B,192,7,7)
// =====================================================================
__global__ void pool_kernel(const float* __restrict__ ctx, float* __restrict__ ctxp)
{
    int idx = blockIdx.x * blockDim.x + threadIdx.x;
    if (idx >= B_ * CHALF * 49) return;
    int l = idx % 49; int t = idx / 49;
    int c = t % CHALF; int b = t / CHALF;
    int i = l / 7, j = l % 7;
    const float* p = ctx + ((long)(b * CHALF + c)) * HW_ + (i * 8) * W_ + j * 8;
    float s = 0.f;
    for (int yy = 0; yy < 8; ++yy)
        for (int xx = 0; xx < 8; ++xx)
            s += p[yy * W_ + xx];
    ctxp[idx] = s * (1.f / 64.f);
}

// =====================================================================
// kf = BN(wk @ ctx_p):  (B, 192, 49)
// =====================================================================
__global__ void kf_kernel(const float* __restrict__ ctxp, const float* __restrict__ wk,
                          const float* __restrict__ gk, const float* __restrict__ bk,
                          float* __restrict__ kf)
{
    int idx = blockIdx.x * blockDim.x + threadIdx.x;
    if (idx >= B_ * CHALF * 49) return;
    int l = idx % 49; int t = idx / 49;
    int o = t % CHALF; int b = t / CHALF;
    float s = 0.f;
    const float* wr = wk + (long)o * CHALF;
    const float* cp = ctxp + (long)b * CHALF * 49 + l;
    for (int c = 0; c < CHALF; ++c) s += wr[c] * cp[c * 49];
    const float invs = rsqrtf(1.f + EPSV);
    kf[idx] = s * (gk[o] * invs) + bk[o];
}

// =====================================================================
// M2[bg, c, o] = sum_l kf[b, g*48+c, l] * wp[o, l]   (o padded to 80)
// =====================================================================
__global__ void m2_kernel(const float* __restrict__ kf, const float* __restrict__ wp,
                          float* __restrict__ M2)
{
    int idx = blockIdx.x * blockDim.x + threadIdx.x;
    if (idx >= B_ * HEADS * CH_ * NPAD) return;
    int o = idx % NPAD; int t = idx / NPAD;
    int c = t % CH_; t /= CH_;
    int g = t % HEADS; int b = t / HEADS;
    float s = 0.f;
    if (o < NTAP) {
        const float* kr = kf + ((long)b * CHALF + g * CH_ + c) * 49;
        const float* wr = wp + (long)o * 49;
        for (int l = 0; l < 49; ++l) s += kr[l] * wr[l];
    }
    M2[idx] = s;
}

// =====================================================================
// dual softmax with closed-form NATTEN RPB, in place
// =====================================================================
DEV int nat_clamp(int t, int k, int dim)
{
    int c = k >> 1;
    if (t < c) return t;
    if (t >= dim - c) return t - (dim - k);
    return c;
}

__global__ void softmax_kernel(float* __restrict__ attn,
                               const float* __restrict__ wpb,
                               const float* __restrict__ rpb1,
                               const float* __restrict__ rpb2)
{
    long idx = (long)blockIdx.x * blockDim.x + threadIdx.x;
    if (idx >= (long)B_ * HEADS * HW_) return;
    int n  = (int)(idx % HW_);
    int bg = (int)(idx / HW_);
    int g  = bg & (HEADS - 1);
    int y  = n / W_, xp = n % W_;
    float* row = attn + idx * NPAD;

    {   // small window (5x5), taps [0,25)
        const int k = KSM, rs = 2 * KSM - 1;
        int by = nat_clamp(H_ - 1 - y, k, H_);
        int bx = nat_clamp(W_ - 1 - xp, k, W_);
        const float* rp = rpb1 + g * rs * rs;
        float a[KSM * KSM];
        float mx = -1e30f;
        #pragma unroll
        for (int i = 0; i < k; ++i)
            #pragma unroll
            for (int j = 0; j < k; ++j) {
                int t = i * k + j;
                float v = row[t] + wpb[t] + rp[(by + i) * rs + (bx + j)];
                a[t] = v; mx = fmaxf(mx, v);
            }
        float sum = 0.f;
        #pragma unroll
        for (int t = 0; t < KSM * KSM; ++t) { a[t] = __expf(a[t] - mx); sum += a[t]; }
        float r = 1.f / sum;
        #pragma unroll
        for (int t = 0; t < KSM * KSM; ++t) row[t] = a[t] * r;
    }
    {   // large window (7x7), taps [25,74)
        const int k = KBIG, rs = 2 * KBIG - 1;
        int by = nat_clamp(H_ - 1 - y, k, H_);
        int bx = nat_clamp(W_ - 1 - xp, k, W_);
        const float* rp = rpb2 + g * rs * rs;
        float a[KBIG * KBIG];
        float mx = -1e30f;
        #pragma unroll
        for (int i = 0; i < k; ++i)
            #pragma unroll
            for (int j = 0; j < k; ++j) {
                int t = i * k + j;
                float v = row[25 + t] + wpb[25 + t] + rp[(by + i) * rs + (bx + j)];
                a[t] = v; mx = fmaxf(mx, v);
            }
        float sum = 0.f;
        #pragma unroll
        for (int t = 0; t < KBIG * KBIG; ++t) { a[t] = __expf(a[t] - mx); sum += a[t]; }
        float r = 1.f / sum;
        #pragma unroll
        for (int t = 0; t < KBIG * KBIG; ++t) row[25 + t] = a[t] * r;
    }
}

// =====================================================================
// Neighborhood AV; output written DIRECTLY in packed-B f16 layout for
// the dp GEMM (k = channel, n = pixel; NT=196, KT=12).
// =====================================================================
__global__ void av_kernel(const float* __restrict__ x, const float* __restrict__ attn,
                          _Float16* __restrict__ outP)
{
    long idx = (long)blockIdx.x * blockDim.x + threadIdx.x;
    if (idx >= (long)B_ * DIM_ * HW_) return;
    int xp = (int)(idx % W_); long t = idx / W_;
    int yp = (int)(t % H_);  t /= H_;
    int ch = (int)(t % DIM_); int b = (int)(t / DIM_);

    int m = ch / CHALF;
    int g = (ch % CHALF) / CH_;
    const int k   = m ? KBIG : KSM;
    const int pad = k / 2;
    const float* arow = attn + (((long)(b * HEADS + g)) * HW_ + yp * W_ + xp) * NPAD + (m ? 25 : 0);
    const float* xc   = x + ((long)(b * DIM_ + ch)) * HW_;

    float s = 0.f;
    for (int i = 0; i < k; ++i) {
        int yy = yp + i - pad;
        if (yy < 0 || yy >= H_) continue;
        for (int j = 0; j < k; ++j) {
            int xx = xp + j - pad;
            if (xx < 0 || xx >= W_) continue;
            s += arow[i * k + j] * xc[yy * W_ + xx];
        }
    }
    // packed-B index for dp GEMM
    int n  = yp * W_ + xp;
    int kt = ch >> 5, kl = ch & 31;
    int hlf = kl >> 4, elem = kl & 15;
    int nt = n >> 4, l16 = n & 15;
    long dst = ((((long)b * 196 + nt) * 12 + kt) * 32 + (hlf << 4) + l16) * 16 + elem;
    outP[dst] = (_Float16)s;
}

// =====================================================================
// host-side orchestration
// =====================================================================
extern "C" void kernel_launch(void* const* d_in, const int* in_sizes, int n_in,
                              void* d_out, int out_size, void* d_ws, size_t ws_size,
                              hipStream_t stream)
{
    (void)in_sizes; (void)n_in; (void)out_size; (void)ws_size;

    const float* x    = (const float*)d_in[0];
    const float* ctx  = (const float*)d_in[1];
    const float* lkw  = (const float*)d_in[2];
    const float* bn0g = (const float*)d_in[3];
    const float* bn0b = (const float*)d_in[4];
    const float* dw0  = (const float*)d_in[5];
    const float* bn1g = (const float*)d_in[6];
    const float* bn1b = (const float*)d_in[7];
    const float* dw1  = (const float*)d_in[8];
    const float* bn2g = (const float*)d_in[9];
    const float* bn2b = (const float*)d_in[10];
    const float* dw2  = (const float*)d_in[11];
    const float* bn3g = (const float*)d_in[12];
    const float* bn3b = (const float*)d_in[13];
    const float* dw3  = (const float*)d_in[14];
    const float* bn4g = (const float*)d_in[15];
    const float* bn4b = (const float*)d_in[16];
    const float* bnog = (const float*)d_in[17];
    const float* bnob = (const float*)d_in[18];
    const float* wq   = (const float*)d_in[19];
    const float* wqg  = (const float*)d_in[20];
    const float* wqb  = (const float*)d_in[21];
    const float* wk   = (const float*)d_in[22];
    const float* wkg  = (const float*)d_in[23];
    const float* wkb  = (const float*)d_in[24];
    const float* wp   = (const float*)d_in[25];
    const float* wpb  = (const float*)d_in[26];
    const float* dpw  = (const float*)d_in[27];
    const float* dpg  = (const float*)d_in[28];
    const float* dpb  = (const float*)d_in[29];
    const float* rpb1 = (const float*)d_in[30];
    const float* rpb2 = (const float*)d_in[31];
    float* out = (float*)d_out;

    // carve workspace
    char* wsb = (char*)d_ws;
    auto carveF = [&](size_t nfloats) -> float* {
        float* p = (float*)wsb;
        wsb += ((nfloats * sizeof(float) + 255) / 256) * 256;
        return p;
    };
    auto carveH = [&](size_t nhalf) -> _Float16* {
        _Float16* p = (_Float16*)wsb;
        wsb += ((nhalf * sizeof(_Float16) + 255) / 256) * 256;
        return p;
    };

    float*     lepe = carveF((size_t)B_ * DIM_ * HW_);           // 77 MB
    float*     qbuf = carveF((size_t)B_ * CHALF * HW_);          // 38 MB
    float*     attn = carveF((size_t)B_ * HEADS * HW_ * NPAD);   // 64 MB
    float*     ctxp = carveF((size_t)B_ * CHALF * 49);
    float*     kfb  = carveF((size_t)B_ * CHALF * 49);
    float*     M2   = carveF((size_t)B_ * HEADS * CH_ * NPAD);
    // packed f16 fragment buffers  [batch][tile][tileK][lane][elem]
    _Float16*  wqP  = carveH((size_t)1  * 12  * 6  * 512);       // wq   192x192
    _Float16*  xP   = carveH((size_t)B_ * 196 * 6  * 512);       // x    192x3136 /batch
    _Float16*  qP   = carveH((size_t)64 * 196 * 2  * 512);       // q^T  3136x48  /bg
    _Float16*  m2P  = carveH((size_t)64 * 8   * 2  * 512);       // M2   48x80    /bg
    _Float16*  dpP  = carveH((size_t)1  * 24  * 12 * 512);       // dp_w 384x384
    _Float16*  avP  = carveH((size_t)B_ * 196 * 12 * 512);       // av   384x3136 /batch

    const float invs  = 1.0f / sqrtf(1.0f + EPSV);               // BN var fold
    const float scale = 1.0f / sqrtf((float)(DIM_ / HEADS));     // HEAD_DIM = 96

    const int TPB = 256;
    auto blk = [&](long n) { return (unsigned)((n + TPB - 1) / TPB); };

    // 1) lepe branch
    lepe_kernel<<<blk((long)B_ * DIM_ * HW_), TPB, 0, stream>>>(
        x, lkw, bn0g, bn0b, dw0, bn1g, bn1b, dw1, bn2g, bn2b,
        dw2, bn3g, bn3b, dw3, bn4g, bn4b, bnog, bnob, lepe);

    // 2) pooling + key features + fused key*weight_proj matrix
    pool_kernel<<<blk(B_ * CHALF * 49), TPB, 0, stream>>>(ctx, ctxp);
    kf_kernel<<<blk(B_ * CHALF * 49), TPB, 0, stream>>>(ctxp, wk, wkg, wkb, kfb);
    m2_kernel<<<blk(B_ * HEADS * CH_ * NPAD), TPB, 0, stream>>>(kfb, wp, M2);

    // 3) pack wq (A) and x-top-half (B) as f16 fragments
    {
        long tot = (long)1 * 12 * 6 * 512;
        pack_a_kernel<<<blk(tot), TPB, 0, stream>>>(wq, CHALF, 1, 0, wqP, CHALF, CHALF, 12, 6, tot);
    }
    {
        long tot = (long)B_ * 196 * 6 * 512;
        pack_b_kernel<<<blk(tot), TPB, 0, stream>>>(x, HW_, 1, (long)DIM_ * HW_, xP,
                                                    HW_, CHALF, 196, 6, tot);
    }

    // 4) q = BN(wq @ x_half) * SCALE     [WMMA, M=192 N=3136 K=192]
    {
        dim3 grid(196 / 4, CHALF / 64, B_);
        wmma_gemm_packed<<<grid, 64, 0, stream>>>(
            wqP, 0, xP, (long)196 * 6, 6,
            qbuf, HW_, 1, (long)CHALF * HW_,
            wqg, wqb, invs, scale,
            nullptr, 0, CHALF, HW_);
    }

    // 5) pack q^T (A, per b*head, K=48 zero-padded to 64) and M2 (B, N=80 -> 8 tiles)
    {
        long tot = (long)64 * 196 * 2 * 512;
        pack_a_kernel<<<blk(tot), TPB, 0, stream>>>(qbuf, 1, HW_, (long)CH_ * HW_, qP,
                                                    HW_, CH_, 196, 2, tot);
    }
    {
        long tot = (long)64 * 8 * 2 * 512;
        pack_b_kernel<<<blk(tot), TPB, 0, stream>>>(M2, NPAD, 1, (long)CH_ * NPAD, m2P,
                                                    NPAD, CH_, 8, 2, tot);
    }

    // 6) attn_pre[bg] = q[bg]^T @ M2[bg]  [WMMA, M=3136 N=80 K=48]
    {
        dim3 grid(8 / 4, HW_ / 64, B_ * HEADS);
        wmma_gemm_packed<<<grid, 64, 0, stream>>>(
            qP, (long)196 * 2, m2P, (long)8 * 2, 2,
            attn, NPAD, 1, (long)HW_ * NPAD,
            nullptr, nullptr, 1.f, 1.f,
            nullptr, 0, HW_, NPAD);
    }

    // 7) + wp_b + RPB, dual softmax (in place)
    softmax_kernel<<<blk((long)B_ * HEADS * HW_), TPB, 0, stream>>>(attn, wpb, rpb1, rpb2);

    // 8) neighborhood AV -> packed-B f16 for dp GEMM
    av_kernel<<<blk((long)B_ * DIM_ * HW_), TPB, 0, stream>>>(x, attn, avP);

    // 9) pack dp_w (A)
    {
        long tot = (long)1 * 24 * 12 * 512;
        pack_a_kernel<<<blk(tot), TPB, 0, stream>>>(dpw, DIM_, 1, 0, dpP, DIM_, DIM_, 24, 12, tot);
    }

    // 10) out = BN(dp_w @ av) + lepe      [WMMA, M=384 N=3136 K=384]
    {
        dim3 grid(196 / 4, DIM_ / 64, B_);
        wmma_gemm_packed<<<grid, 64, 0, stream>>>(
            dpP, 0, avP, (long)196 * 12, 12,
            out, HW_, 1, (long)DIM_ * HW_,
            dpg, dpb, invs, 1.f,
            lepe, (long)DIM_ * HW_,
            DIM_, HW_);
    }
}
